// SBOHead_74440373174329
// MI455X (gfx1250) — matrix operand order repaired and checked by reading, not compiled, not run on hardware
//
#include <hip/hip_runtime.h>

// ---------------------------------------------------------------------------
// SBO head for MI455X (gfx1250, wave32). All four GEMMs run on
// v_wmma_f32_16x16x32_bf16 (f32 accumulate) with a double-buffered LDS
// pipeline fed by GLOBAL_LOAD_ASYNC_TO_LDS_B128 (ASYNCcnt-tracked DMA).
// ~432 GFLOP vs ~0.8 GB HBM traffic -> compute-bound on the matrix pipe.
// Steady-state K-loop is branch-free: wait_asynccnt(4) / barrier / ds_load
// fragments / barrier / prefetch k+2 / 8x WMMA. Last two K-steps are peeled.
// ---------------------------------------------------------------------------

#define BATCH   2
#define SEQ     2048
#define DIM     1024
#define FFN_DIM 4096
#define VOCAB   32000
#define MAXSPAN 10
#define BS      (BATCH * SEQ)          // 4096 token rows
#define D3      (3 * DIM)              // 3072
#define LN_EPS  1e-5f

typedef __attribute__((ext_vector_type(16))) __bf16 v16bf;
typedef __attribute__((ext_vector_type(8)))  float  v8f;

union FragAB { v16bf v; unsigned u[8]; };
union FragC  { v8f   v; float    f[8]; };

// round-to-nearest-even f32 -> bf16 (as raw u16)
static __device__ __forceinline__ unsigned short f32_to_bf16(float f) {
    unsigned u = __float_as_uint(f);
    u += 0x7FFFu + ((u >> 16) & 1u);
    return (unsigned short)(u >> 16);
}
static __device__ __forceinline__ float gelu_exact(float x) {
    return 0.5f * x * (1.0f + erff(x * 0.70710678118654752f));
}

// 32-bit LDS offset of a generic pointer into __shared__
static __device__ __forceinline__ unsigned lds_off_u32(const void* p) {
    return (unsigned)(unsigned long long)(__attribute__((address_space(3))) const char*)p;
}
// CDNA5 async DMA: global (16B/lane) -> LDS, tracked by ASYNCcnt
static __device__ __forceinline__ void async_load_b128(const unsigned short* g, unsigned lds_off) {
    asm volatile("global_load_async_to_lds_b128 %0, %1, off"
                 :: "v"(lds_off), "v"((unsigned long long)g)
                 : "memory");
}
static __device__ __forceinline__ void wait_asynccnt_le4() {
    asm volatile("s_wait_asynccnt 0x4" ::: "memory");
}
static __device__ __forceinline__ void wait_asynccnt_0() {
    asm volatile("s_wait_asynccnt 0x0" ::: "memory");
}

// ---------------------------------------------------------------------------
// f32 -> bf16 conversion (memory-bound, trivial)
// ---------------------------------------------------------------------------
__global__ __launch_bounds__(256) void cvt_f32_bf16(const float* __restrict__ src,
                                                    unsigned short* __restrict__ dst,
                                                    int n) {
    int stride = gridDim.x * 256;
    for (int i = blockIdx.x * 256 + threadIdx.x; i < n; i += stride)
        dst[i] = f32_to_bf16(src[i]);
}

// ---------------------------------------------------------------------------
// Span metadata: per-batch cummax (span start) / reverse cummin (span end)
// via Hillis-Steele scans in LDS. One block per batch row.
// ---------------------------------------------------------------------------
__global__ __launch_bounds__(256) void span_meta_kernel(const int* __restrict__ mask,
                                                        int* __restrict__ start_idx,
                                                        int* __restrict__ end_idx,
                                                        int* __restrict__ pos_out) {
    __shared__ int sm[SEQ];
    __shared__ int buf[2][SEQ];
    const int b = blockIdx.x;
    const int tid = threadIdx.x;

    for (int s = tid; s < SEQ; s += 256) sm[s] = mask[b * SEQ + s];
    __syncthreads();

    // forward cummax of (is_start ? s : -1)
    for (int s = tid; s < SEQ; s += 256) {
        int m = sm[s];
        int prev = (s > 0) ? sm[s - 1] : 0;
        buf[0][s] = (m == 1 && prev != 1) ? s : -1;
    }
    __syncthreads();
    int ping = 0;
    for (int step = 1; step < SEQ; step <<= 1) {
        for (int s = tid; s < SEQ; s += 256) {
            int v = buf[ping][s];
            if (s >= step) { int w = buf[ping][s - step]; v = (v > w) ? v : w; }
            buf[1 - ping][s] = v;
        }
        ping ^= 1;
        __syncthreads();
    }
    for (int s = tid; s < SEQ; s += 256) {
        int st = buf[ping][s];
        start_idx[b * SEQ + s] = st;
        int p = s - st;
        p = (p < 0) ? 0 : ((p > 3 * MAXSPAN - 1) ? 3 * MAXSPAN - 1 : p);
        pos_out[b * SEQ + s] = p;
    }
    __syncthreads();

    // backward cummin of (is_end ? s : SEQ)
    for (int s = tid; s < SEQ; s += 256) {
        int m = sm[s];
        int nxt = (s < SEQ - 1) ? sm[s + 1] : 0;
        buf[0][s] = (m == 1 && nxt != 1) ? s : SEQ;
    }
    __syncthreads();
    ping = 0;
    for (int step = 1; step < SEQ; step <<= 1) {
        for (int s = tid; s < SEQ; s += 256) {
            int v = buf[ping][s];
            if (s + step < SEQ) { int w = buf[ping][s + step]; v = (v < w) ? v : w; }
            buf[1 - ping][s] = v;
        }
        ping ^= 1;
        __syncthreads();
    }
    for (int s = tid; s < SEQ; s += 256) end_idx[b * SEQ + s] = buf[ping][s];
}

// ---------------------------------------------------------------------------
// Overwrite masked rows of h with [ctx_start | pos_emb | ctx_end] (bf16)
// ---------------------------------------------------------------------------
__global__ __launch_bounds__(256) void scatter_h0_kernel(
    const unsigned short* __restrict__ hs_bf16,   // [BS, DIM]
    const float* __restrict__ pos_emb,            // [3*MAXSPAN, DIM]
    const int* __restrict__ mask,
    const int* __restrict__ start_idx,
    const int* __restrict__ end_idx,
    const int* __restrict__ pos_in,
    unsigned short* __restrict__ h)               // [BS, 3*DIM]
{
    const int tok = blockIdx.x;
    if (mask[tok] != 1) return;
    const int b = tok / SEQ;
    const int st = start_idx[tok], en = end_idx[tok], p = pos_in[tok];
    int cs = st - 1; cs = (cs < 0) ? 0 : ((cs > SEQ - 1) ? SEQ - 1 : cs);
    int ce = en + 1; ce = (ce < 0) ? 0 : ((ce > SEQ - 1) ? SEQ - 1 : ce);
    const unsigned short* src_s = hs_bf16 + (size_t)(b * SEQ + cs) * DIM;
    const unsigned short* src_e = hs_bf16 + (size_t)(b * SEQ + ce) * DIM;
    const float* pe = pos_emb + (size_t)p * DIM;
    unsigned short* dst = h + (size_t)tok * D3;
    for (int j = threadIdx.x; j < DIM; j += 256) {
        dst[j]           = src_s[j];
        dst[DIM + j]     = f32_to_bf16(pe[j]);
        dst[2 * DIM + j] = src_e[j];
    }
}

// ---------------------------------------------------------------------------
// Row LayerNorm: f32 in -> bf16 out (biased variance, matches jnp.var)
// ---------------------------------------------------------------------------
__global__ __launch_bounds__(256) void ln_f32_to_bf16(const float* __restrict__ in,
                                                      const float* __restrict__ g,
                                                      const float* __restrict__ be,
                                                      unsigned short* __restrict__ out,
                                                      int N) {
    const int row = blockIdx.x;
    const int tid = threadIdx.x;
    const float* x = in + (size_t)row * N;
    float s = 0.f, s2 = 0.f;
    for (int j = tid; j < N; j += 256) { float v = x[j]; s += v; s2 += v * v; }
    __shared__ float r0[256], r1[256];
    r0[tid] = s; r1[tid] = s2;
    __syncthreads();
    for (int off = 128; off > 0; off >>= 1) {
        if (tid < off) { r0[tid] += r0[tid + off]; r1[tid] += r1[tid + off]; }
        __syncthreads();
    }
    const float mu = r0[0] / (float)N;
    const float var = r1[0] / (float)N - mu * mu;
    const float rstd = rsqrtf(var + LN_EPS);
    unsigned short* o = out + (size_t)row * N;
    for (int j = tid; j < N; j += 256)
        o[j] = f32_to_bf16((x[j] - mu) * rstd * g[j] + be[j]);
}

// ---------------------------------------------------------------------------
// Tiled bf16 WMMA GEMM:  C[M,N] = act(A[M,K] @ W[N,K]^T + bias)
// Block = 256 thr (8 waves), tile 128x128, K-step 32, double-buffered LDS
// fed by global_load_async_to_lds_b128 (4 async instr / wave / tile).
// Each wave: 32x64 C region = 2x4 tiles -> 8 v_wmma per K-step.
// LDS row stride 40 elems (80B) -> conflict-free b128 fragment gathers.
// ---------------------------------------------------------------------------
#define LDSS 40

template <bool HAS_BIAS, bool GELU, bool OUT_BF16>
__global__ __launch_bounds__(256) void gemm_bf16_wmma(
    const unsigned short* __restrict__ A,   // [M,K] bf16 row-major
    const unsigned short* __restrict__ W,   // [N,K] bf16 row-major
    const float* __restrict__ bias,         // [N] or null
    void* __restrict__ outp,                // [M,N] bf16 or f32
    int M, int N, int K)
{
    __shared__ __align__(16) unsigned short sA[2][128 * LDSS];
    __shared__ __align__(16) unsigned short sB[2][128 * LDSS];

    const int tid  = threadIdx.x;
    const int lane = tid & 31;
    const int wave = tid >> 5;
    const int wm = wave & 3;         // 4 waves along M
    const int wn = wave >> 2;        // 2 waves along N
    const int m0 = wm * 32, n0 = wn * 64;
    const int fm = lane & 15, half = lane >> 4;

    const int rowBase = blockIdx.y * 128;
    const int colBase = blockIdx.x * 128;

    // async-copy assignment: each thread moves one 16B chunk of 2 rows of A
    // and 2 rows of W per K-step (4 async instructions per wave per tile)
    const int lr = tid >> 2;             // 0..63
    const int lc = (tid & 3) << 3;       // 0,8,16,24
    const unsigned short* gA0 = A + (size_t)(rowBase + lr)      * K + lc;
    const unsigned short* gA1 = A + (size_t)(rowBase + lr + 64) * K + lc;
    const unsigned short* gB0 = W + (size_t)(colBase + lr)      * K + lc;
    const unsigned short* gB1 = W + (size_t)(colBase + lr + 64) * K + lc;
    unsigned dA0[2], dA1[2], dB0[2], dB1[2];
#pragma unroll
    for (int b = 0; b < 2; ++b) {
        dA0[b] = lds_off_u32(&sA[b][lr * LDSS + lc]);
        dA1[b] = lds_off_u32(&sA[b][(lr + 64) * LDSS + lc]);
        dB0[b] = lds_off_u32(&sB[b][lr * LDSS + lc]);
        dB1[b] = lds_off_u32(&sB[b][(lr + 64) * LDSS + lc]);
    }
    auto issue_tile = [&](int buf, int k0) {
        async_load_b128(gA0 + k0, dA0[buf]);
        async_load_b128(gA1 + k0, dA1[buf]);
        async_load_b128(gB0 + k0, dB0[buf]);
        async_load_b128(gB1 + k0, dB1[buf]);
    };

    FragC acc[2][4];
#pragma unroll
    for (int i = 0; i < 2; ++i)
#pragma unroll
        for (int j = 0; j < 4; ++j)
#pragma unroll
            for (int p = 0; p < 8; ++p) acc[i][j].f[p] = 0.f;

    auto load_frags = [&](int cur, FragAB (&af)[2], FragAB (&bfr)[4]) {
#pragma unroll
        for (int i2 = 0; i2 < 2; ++i2) {
            const unsigned short* base = &sA[cur][(m0 + i2 * 16 + fm) * LDSS];
#pragma unroll
            for (int i = 0; i < 8; ++i) {
                // ISA A-layout (16x32 bf16): VGPR i<4 -> K=2i(+8*half), i>=4 -> K=16+2(i-4)(+8*half)
                const int kk = ((i < 4) ? (2 * i) : (2 * i + 8)) + 8 * half;
                af[i2].u[i] = *(const unsigned*)(base + kk);
            }
        }
#pragma unroll
        for (int j = 0; j < 4; ++j) {
            const unsigned short* base = &sB[cur][(n0 + j * 16 + fm) * LDSS];
#pragma unroll
            for (int i = 0; i < 8; ++i) {
                // B-layout matches A: VGPR i<4 -> K=2i(+8*half), i>=4 -> K=16+2(i-4)(+8*half)
                const int kk = ((i < 4) ? (2 * i) : (2 * i + 8)) + 8 * half;
                bfr[j].u[i] = *(const unsigned*)(base + kk);
            }
        }
    };
    auto run_wmma = [&](FragAB (&af)[2], FragAB (&bfr)[4]) {
#pragma unroll
        for (int i2 = 0; i2 < 2; ++i2)
#pragma unroll
            for (int j = 0; j < 4; ++j)
                acc[i2][j].v = __builtin_amdgcn_wmma_f32_16x16x32_bf16(
                    false, af[i2].v, false, bfr[j].v,
                    (short)0, acc[i2][j].v, false, false);
    };

    const int nk = K >> 5;
    issue_tile(0, 0);
    if (nk > 1) issue_tile(1, 32);

    int kt = 0;
    // steady state: two tiles in flight, branch-free body
    for (; kt + 2 < nk; ++kt) {
        const int cur = kt & 1;
        wait_asynccnt_le4();             // drain the older tile (in-order)
        __syncthreads();
        FragAB af[2], bfr[4];
        load_frags(cur, af, bfr);
        __syncthreads();                 // buf[cur] fully read by all waves
        issue_tile(cur, (kt + 2) << 5);  // prefetch overlaps the WMMAs below
        run_wmma(af, bfr);
    }
    if (nk > 1) {                        // kt == nk-2: one tile still behind it
        const int cur = kt & 1;
        wait_asynccnt_le4();
        __syncthreads();
        FragAB af[2], bfr[4];
        load_frags(cur, af, bfr);
        run_wmma(af, bfr);
        ++kt;
        __syncthreads();
    }
    {                                    // kt == nk-1: last tile
        const int cur = kt & 1;
        wait_asynccnt_0();
        __syncthreads();
        FragAB af[2], bfr[4];
        load_frags(cur, af, bfr);
        run_wmma(af, bfr);
    }

    float* outF = (float*)outp;
    unsigned short* outB = (unsigned short*)outp;
#pragma unroll
    for (int i2 = 0; i2 < 2; ++i2) {
#pragma unroll
        for (int j = 0; j < 4; ++j) {
            const int col = colBase + n0 + j * 16 + fm;
            const float bv = HAS_BIAS ? bias[col] : 0.f;
#pragma unroll
            for (int p = 0; p < 8; ++p) {
                const int row = rowBase + m0 + i2 * 16 + p + 8 * half;
                float v = acc[i2][j].f[p] + bv;
                if (GELU) v = gelu_exact(v);
                const size_t off = (size_t)row * (size_t)N + (size_t)col;
                if (OUT_BF16) outB[off] = f32_to_bf16(v);
                else          outF[off] = v;
            }
        }
    }
}

// ---------------------------------------------------------------------------
// Host launcher
// ---------------------------------------------------------------------------
extern "C" void kernel_launch(void* const* d_in, const int* in_sizes, int n_in,
                              void* d_out, int out_size, void* d_ws, size_t ws_size,
                              hipStream_t stream) {
    const float* hs     = (const float*)d_in[0];   // [B,S,D]
    const int*   mask   = (const int*)  d_in[1];   // [B,S]
    const float* Wproj  = (const float*)d_in[2];   // [3D,D]
    const float* posEmb = (const float*)d_in[3];   // [3*MAXSPAN,D]
    const float* W1     = (const float*)d_in[4];   // [FFN,3D]
    const float* b1     = (const float*)d_in[5];
    const float* g1     = (const float*)d_in[6];
    const float* be1    = (const float*)d_in[7];
    const float* W2     = (const float*)d_in[8];   // [D,FFN]
    const float* g2     = (const float*)d_in[9];
    const float* be2    = (const float*)d_in[10];
    const float* Wcls   = (const float*)d_in[11];  // [V,D]
    const float* bcls   = (const float*)d_in[12];
    float* out = (float*)d_out;                    // [B,S,V]

    char* base = (char*)d_ws;
    size_t off = 0;
    auto alloc = [&](size_t bytes) -> char* {
        char* p = base + off;
        off += (bytes + 255) & ~(size_t)255;
        return p;
    };

    unsigned short* hs_bf   = (unsigned short*)alloc((size_t)BS * DIM * 2);
    unsigned short* wp_bf   = (unsigned short*)alloc((size_t)D3 * DIM * 2);
    unsigned short* w1_bf   = (unsigned short*)alloc((size_t)FFN_DIM * D3 * 2);
    unsigned short* w2_bf   = (unsigned short*)alloc((size_t)DIM * FFN_DIM * 2);
    unsigned short* wc_bf   = (unsigned short*)alloc((size_t)VOCAB * DIM * 2);
    unsigned short* h_bf    = (unsigned short*)alloc((size_t)BS * D3 * 2);
    float*          t_raw   = (float*)         alloc((size_t)BS * FFN_DIM * 4);
    unsigned short* t_bf    = (unsigned short*)alloc((size_t)BS * FFN_DIM * 2);
    int*            startIx = (int*)           alloc((size_t)BS * 4);
    int*            endIx   = (int*)           alloc((size_t)BS * 4);
    int*            posIx   = (int*)           alloc((size_t)BS * 4);
    // aliases (non-overlapping lifetimes):
    float*          z_raw   = t_raw;      // t_raw dead after LN1
    unsigned short* z_bf    = hs_bf;      // hs_bf dead after scatter

    auto cvt = [&](const float* s, unsigned short* d, size_t n) {
        int grid = (int)((n + 256 * 8 - 1) / (256 * 8));
        if (grid > 4096) grid = 4096;
        cvt_f32_bf16<<<grid, 256, 0, stream>>>(s, d, (int)n);
    };
    cvt(hs,    hs_bf, (size_t)BS * DIM);
    cvt(Wproj, wp_bf, (size_t)D3 * DIM);
    cvt(W1,    w1_bf, (size_t)FFN_DIM * D3);
    cvt(W2,    w2_bf, (size_t)DIM * FFN_DIM);
    cvt(Wcls,  wc_bf, (size_t)VOCAB * DIM);

    span_meta_kernel<<<BATCH, 256, 0, stream>>>(mask, startIx, endIx, posIx);

    // GEMM1: h = hs @ Wproj^T            [4096,1024]x[3072,1024]^T -> bf16 h
    gemm_bf16_wmma<false, false, true><<<dim3(D3 / 128, BS / 128), 256, 0, stream>>>(
        hs_bf, wp_bf, nullptr, h_bf, BS, D3, DIM);

    scatter_h0_kernel<<<BS, 256, 0, stream>>>(hs_bf, posEmb, mask, startIx, endIx, posIx, h_bf);

    // GEMM2: t_raw = gelu(h @ W1^T + b1) [4096,3072]x[4096,3072]^T -> f32
    gemm_bf16_wmma<true, true, false><<<dim3(FFN_DIM / 128, BS / 128), 256, 0, stream>>>(
        h_bf, w1_bf, b1, t_raw, BS, FFN_DIM, D3);

    ln_f32_to_bf16<<<BS, 256, 0, stream>>>(t_raw, g1, be1, t_bf, FFN_DIM);

    // GEMM3: z_raw = gelu(t @ W2^T)      [4096,4096]x[1024,4096]^T -> f32
    gemm_bf16_wmma<false, true, false><<<dim3(DIM / 128, BS / 128), 256, 0, stream>>>(
        t_bf, w2_bf, nullptr, z_raw, BS, DIM, FFN_DIM);

    ln_f32_to_bf16<<<BS, 256, 0, stream>>>(z_raw, g2, be2, z_bf, DIM);

    // GEMM4: out = z @ Wcls^T + bcls     [4096,1024]x[32000,1024]^T -> f32 out
    gemm_bf16_wmma<true, false, false><<<dim3(VOCAB / 128, BS / 128), 256, 0, stream>>>(
        z_bf, wc_bf, bcls, out, BS, VOCAB, DIM);
}
